// DeltaBlock_54150947668398
// MI455X (gfx1250) — compile-verified, matrix operand order checked by reading, not run on hardware
//
#include <hip/hip_runtime.h>
#include <cmath>
#include <cstdint>

// ---------------- problem constants ----------------
constexpr int   kD  = 1024;
constexpr int   kB  = 4;
constexpr int   kL  = 4096;
constexpr int   kC  = 64;                  // chunk
constexpr int   kN  = kL / kC;             // 64 chunks per sequence
constexpr float kAlpha = 1.0f;
constexpr int   kML = kB * kL;             // 16384 rows total

// ---------------- optional CDNA5 async-to-LDS path (guarded) ----------------
#if defined(__has_builtin)
#if __has_builtin(__builtin_amdgcn_global_load_async_to_lds_b128) && \
    __has_builtin(__builtin_amdgcn_s_wait_asynccnt)
#define USE_ASYNC_LDS 1
#endif
#endif

#ifdef USE_ASYNC_LDS
// clang prototype is (v4i addrspace(1)*, v4i addrspace(3)*, imm offset, imm cpol)
typedef int v4i_vs __attribute__((vector_size(16)));
typedef __attribute__((address_space(1))) v4i_vs* gptr128_t;
typedef __attribute__((address_space(3))) v4i_vs* lptr128_t;
#endif

// ---------------- WMMA plumbing (CDNA5 / gfx1250, wave32) ----------------
typedef __attribute__((ext_vector_type(16))) __bf16 bf16x16;
typedef __attribute__((ext_vector_type(8)))  float  f32x8;

struct __align__(16) U4 { unsigned u0, u1, u2, u3; };

union Frag {
  bf16x16 v;
  U4      q[2];
  __bf16  e[16];
};

__device__ __forceinline__ f32x8 fzero() {
  f32x8 r;
  for (int i = 0; i < 8; ++i) r[i] = 0.0f;
  return r;
}

// One 16x32 A-fragment / 32x16 B-fragment from a K-contiguous row.
// Two 16B reads at byte offsets khalf*16 and 32+khalf*16; rows 16B-aligned.
__device__ __forceinline__ bf16x16 frag_row(const __bf16* rowk, int khalf) {
  Frag f;
  const char* p = (const char*)rowk;
  f.q[0] = *(const U4*)(p + khalf * 16);
  f.q[1] = *(const U4*)(p + 32 + khalf * 16);
  return f.v;
}

__device__ __forceinline__ f32x8 wmma_bf16(bf16x16 a, bf16x16 b, f32x8 c) {
  return __builtin_amdgcn_wmma_f32_16x16x32_bf16(
      false, a, false, b, (short)0, c, false, false);
}

// ---------------- generic helpers ----------------
__global__ void k_cvt_bf16(const float* __restrict__ src, __bf16* __restrict__ dst, int n) {
  int i = blockIdx.x * blockDim.x + threadIdx.x;
  int stride = gridDim.x * blockDim.x;
  for (; i < n; i += stride) dst[i] = (__bf16)src[i];
}

__global__ void k_zeroS(float* __restrict__ Sf, __bf16* __restrict__ Sb, int n) {
  int i = blockIdx.x * blockDim.x + threadIdx.x;
  int stride = gridDim.x * blockDim.x;
  for (; i < n; i += stride) { Sf[i] = 0.0f; Sb[i] = (__bf16)0.0f; }
}

// beta = sigmoid(X @ beta_w^T + beta_b); one wave per row.
__global__ __launch_bounds__(256)
void k_beta(const float* __restrict__ X, const float* __restrict__ bw,
            const float* __restrict__ bb, float* __restrict__ beta) {
  int wid = threadIdx.x >> 5, lane = threadIdx.x & 31;
  int row = blockIdx.x * 8 + wid;
  const float* xr = X + (size_t)row * kD;
  float s = 0.0f;
  for (int j = lane; j < kD; j += 32) s += xr[j] * bw[j];
  for (int off = 16; off > 0; off >>= 1) s += __shfl_down(s, off, 32);
  if (lane == 0) beta[row] = kAlpha / (1.0f + expf(-(s + bb[0])));
}

// ---------------- QKV projection (NT GEMM, direct fragment loads) ----------------
// z=0: Q.  z=1: K, Kb=K*beta (row-major) + KT, KbT (transposed [b][d][l]).
// z=2: VbT = (V/alpha)*beta transposed only.
__global__ __launch_bounds__(256)
void k_gemm_qkv(const __bf16* __restrict__ XB,
                const __bf16* __restrict__ WQ, const __bf16* __restrict__ WK,
                const __bf16* __restrict__ WV,
                const float* __restrict__ bq, const float* __restrict__ bk,
                const float* __restrict__ bv, const float* __restrict__ beta,
                __bf16* __restrict__ Q, __bf16* __restrict__ K,
                __bf16* __restrict__ Kb, __bf16* __restrict__ KT,
                __bf16* __restrict__ KbT, __bf16* __restrict__ VbT) {
  const int tid = threadIdx.x, lane = tid & 31, w = tid >> 5;
  const int khalf = lane >> 4, lm = lane & 15;
  const int n0 = blockIdx.x * 128, m0 = blockIdx.y * 128, z = blockIdx.z;
  const __bf16* Wmat = (z == 0) ? WQ : (z == 1) ? WK : WV;
  const float* bias = (z == 0) ? bq : (z == 1) ? bk : bv;
  const int wm = (w & 3) * 32, wn = (w >> 2) * 64;

  f32x8 acc[2][4];
  for (int i = 0; i < 2; ++i) for (int j = 0; j < 4; ++j) acc[i][j] = fzero();

  for (int kk = 0; kk < kD; kk += 32) {
    bf16x16 a[2], b[4];
    for (int i = 0; i < 2; ++i) {
      const __bf16* ar = XB + (size_t)(m0 + wm + i * 16 + lm) * kD + kk;
      __builtin_prefetch(ar + 128, 0, 3);
      a[i] = frag_row(ar, khalf);
    }
    for (int j = 0; j < 4; ++j)
      b[j] = frag_row(Wmat + (size_t)(n0 + wn + j * 16 + lm) * kD + kk, khalf);
    for (int i = 0; i < 2; ++i)
      for (int j = 0; j < 4; ++j)
        acc[i][j] = wmma_bf16(a[i], b[j], acc[i][j]);
  }

  for (int i = 0; i < 2; ++i)
    for (int r = 0; r < 8; ++r) {
      int m = m0 + wm + i * 16 + r + 8 * khalf;
      float be = beta[m];
      int b0 = m >> 12;            // sequence index (kL = 4096)
      int l  = m & (kL - 1);       // position within sequence
      for (int j = 0; j < 4; ++j) {
        int n = n0 + wn + j * 16 + lm;
        float v = acc[i][j][r] + bias[n];
        size_t idx  = (size_t)m * kD + n;
        size_t tidx = ((size_t)b0 * kD + n) * kL + l;
        if (z == 0) {
          Q[idx] = (__bf16)v;
        } else if (z == 1) {
          __bf16 kv = (__bf16)v, kbv = (__bf16)(v * be);
          K[idx]  = kv;  Kb[idx]  = kbv;
          KT[tidx] = kv; KbT[tidx] = kbv;
        } else {
          VbT[tidx] = (__bf16)((v / kAlpha) * be);
        }
      }
    }
}

// ---------------- per-chunk prep: T in LDS, W = T@Kb, U = T@Vb ----------------
__global__ __launch_bounds__(256)
void k_chunkprep(const __bf16* __restrict__ K, const __bf16* __restrict__ Kb,
                 const __bf16* __restrict__ KbT, const __bf16* __restrict__ VbT,
                 __bf16* __restrict__ Wm, __bf16* __restrict__ Um) {
  __shared__ float  Ts[64][65];
  __shared__ __bf16 Tb[64][64];
  __shared__ float  red[4][64];

  const int tid = threadIdx.x, lane = tid & 31, w = tid >> 5;
  const int khalf = lane >> 4, lm = lane & 15;
  const int b = blockIdx.x >> 6, cn = blockIdx.x & 63;
  const size_t rowBase = (size_t)blockIdx.x * kC;   // == b*kL + cn*kC
  const int l0 = cn * kC;

  // phase 1: Tfull = Kb_c @ K_c^T  (M=N=64, K=1024)
  for (int t = w; t < 16; t += 8) {
    int tm = t >> 2, tn = t & 3;
    f32x8 acc = fzero();
    const __bf16* arow = Kb + (rowBase + tm * 16 + lm) * kD;
    const __bf16* brow = K  + (rowBase + tn * 16 + lm) * kD;
    for (int kk = 0; kk < kD; kk += 32)
      acc = wmma_bf16(frag_row(arow + kk, khalf), frag_row(brow + kk, khalf), acc);
    for (int r = 0; r < 8; ++r) {
      int m = tm * 16 + r + 8 * khalf, n = tn * 16 + lm;
      Ts[m][n] = (n < m) ? -acc[r] : 0.0f;
    }
  }
  __syncthreads();

  // phase 2: 64 sequential row updates (small: VALU f32 in LDS)
  {
    int c = tid & 63, part = tid >> 6;
    for (int i = 1; i < 64; ++i) {
      float p = 0.0f;
      int mb = part * 16;
      for (int mm = mb; mm < mb + 16; ++mm) p += Ts[i][mm] * Ts[mm][c];
      red[part][c] = p;
      __syncthreads();
      if (tid < 64) Ts[i][tid] += red[0][tid] + red[1][tid] + red[2][tid] + red[3][tid];
      __syncthreads();
    }
  }
  if (tid < 64) Ts[tid][tid] += 1.0f;
  __syncthreads();
  for (int e = tid; e < 64 * 64; e += 256) Tb[e >> 6][e & 63] = (__bf16)Ts[e >> 6][e & 63];
  __syncthreads();

  // phase 3: W = T@Kb, U = T@Vb (B from transposed global buffers, no staging)
  const int tmw = w & 3;
  bf16x16 aT0 = frag_row(&Tb[tmw * 16 + lm][0], khalf);
  bf16x16 aT1 = frag_row(&Tb[tmw * 16 + lm][32], khalf);
  for (int sel = 0; sel < 2; ++sel) {
    const __bf16* srcT = sel ? VbT : KbT;
    __bf16* dst = sel ? Um : Wm;
    for (int j = 0; j < 32; ++j) {
      int nt = (w >> 2) + j * 2;                 // 0..63 n-tiles
      int n = nt * 16 + lm;
      const __bf16* brow = srcT + ((size_t)b * kD + n) * kL + l0;
      f32x8 acc = fzero();
      acc = wmma_bf16(aT0, frag_row(brow, khalf), acc);
      acc = wmma_bf16(aT1, frag_row(brow + 32, khalf), acc);
      for (int r = 0; r < 8; ++r) {
        int m = tmw * 16 + r + 8 * khalf;
        dst[(rowBase + m) * kD + nt * 16 + lm] = (__bf16)acc[r];
      }
    }
  }
}

// ---------------- scan step A: u_i = U_t - W_t @ S ; nb==8: Ab = tril(Q K^T) ----------------
__global__ __launch_bounds__(256)
void k_stepA(const __bf16* __restrict__ Wm, const __bf16* __restrict__ Um,
             const __bf16* __restrict__ Q, const __bf16* __restrict__ K,
             const __bf16* __restrict__ SbT,
             __bf16* __restrict__ UIT, __bf16* __restrict__ AB, int t) {
  const int tid = threadIdx.x, lane = tid & 31, w = tid >> 5;
  const int khalf = lane >> 4, lm = lane & 15;
  const int b = blockIdx.y, nb = blockIdx.x;
  const size_t rowBase = (size_t)b * kL + (size_t)t * kC;

  if (nb < 8) {
    const __bf16* SBb = SbT + (size_t)b * kD * kD;
    const int n0 = nb * 128, tmw = w & 3, tn0 = (w >> 2) * 4;
    f32x8 acc[4];
    for (int j = 0; j < 4; ++j) acc[j] = fzero();

#ifdef USE_ASYNC_LDS
    __shared__ __bf16 Araw[2][64][32];   // double-buffered W tile via async DMA
    auto issueA = [&](int buf, int kk) {
      int row = tid >> 2, seg = (tid & 3) * 8;
      const __bf16* src = Wm + (rowBase + row) * kD + kk + seg;
      __builtin_amdgcn_global_load_async_to_lds_b128(
          (gptr128_t)(uintptr_t)src,
          (lptr128_t)(unsigned)(uintptr_t)&Araw[buf][row][seg], 0, 0);
    };
    auto compute = [&](int cur, int kk) {
      bf16x16 a = frag_row(&Araw[cur][tmw * 16 + lm][0], khalf);
      for (int j = 0; j < 4; ++j) {
        int n = n0 + (tn0 + j) * 16 + lm;
        acc[j] = wmma_bf16(a, frag_row(SBb + (size_t)n * kD + kk, khalf), acc[j]);
      }
    };
    issueA(0, 0);
    int kk = 0;
    for (; kk < kD - 32; kk += 32) {
      int cur = (kk >> 5) & 1;
      issueA(cur ^ 1, kk + 32);
      __builtin_amdgcn_s_wait_asynccnt(1);
      __syncthreads();
      compute(cur, kk);
      __syncthreads();
    }
    __builtin_amdgcn_s_wait_asynccnt(0);
    __syncthreads();
    compute((kk >> 5) & 1, kk);
#else
    for (int kk = 0; kk < kD; kk += 32) {
      bf16x16 a = frag_row(Wm + (rowBase + tmw * 16 + lm) * kD + kk, khalf);
      for (int j = 0; j < 4; ++j) {
        int n = n0 + (tn0 + j) * 16 + lm;
        acc[j] = wmma_bf16(a, frag_row(SBb + (size_t)n * kD + kk, khalf), acc[j]);
      }
    }
#endif

    for (int j = 0; j < 4; ++j) {
      int n = n0 + (tn0 + j) * 16 + lm;
      for (int r = 0; r < 8; ++r) {
        int m = tmw * 16 + r + 8 * khalf;
        float u = (float)Um[(rowBase + m) * kD + n] - acc[j][r];
        UIT[((size_t)b * kD + n) * kC + m] = (__bf16)u;   // transposed, contiguous in r
      }
    }
  } else {
    // Ab = tril(Q_t @ K_t^T) inclusive
    for (int t2 = w; t2 < 16; t2 += 8) {
      int tm = t2 >> 2, tn = t2 & 3;
      f32x8 acc = fzero();
      const __bf16* arow = Q + (rowBase + tm * 16 + lm) * kD;
      const __bf16* brow = K + (rowBase + tn * 16 + lm) * kD;
      for (int kk = 0; kk < kD; kk += 32)
        acc = wmma_bf16(frag_row(arow + kk, khalf), frag_row(brow + kk, khalf), acc);
      for (int r = 0; r < 8; ++r) {
        int m = tm * 16 + r + 8 * khalf, n = tn * 16 + lm;
        AB[((size_t)b * kC + m) * kC + n] = (__bf16)((n <= m) ? acc[r] : 0.0f);
      }
    }
  }
}

// ---------------- scan step B: o = Ab @ u_i + Q_t @ S (no LDS, no barriers) ----------------
__global__ __launch_bounds__(256)
void k_stepB(const __bf16* __restrict__ AB, const __bf16* __restrict__ UIT,
             const __bf16* __restrict__ Q, const __bf16* __restrict__ SbT,
             __bf16* __restrict__ OB, int t) {
  const int tid = threadIdx.x, lane = tid & 31, w = tid >> 5;
  const int khalf = lane >> 4, lm = lane & 15;
  const int b = blockIdx.y, nb = blockIdx.x, n0 = nb * 128;
  const int tmw = w & 3, tn0 = (w >> 2) * 4;
  const size_t rowBase = (size_t)b * kL + (size_t)t * kC;
  const __bf16* SBb = SbT + (size_t)b * kD * kD;

  f32x8 acc[4];
  for (int j = 0; j < 4; ++j) acc[j] = fzero();

  // part 1: Ab @ u_i (K = 64)
  const __bf16* arowAB = AB + ((size_t)b * kC + tmw * 16 + lm) * kC;
  bf16x16 aA0 = frag_row(arowAB, khalf);
  bf16x16 aA1 = frag_row(arowAB + 32, khalf);
  for (int j = 0; j < 4; ++j) {
    int n = n0 + (tn0 + j) * 16 + lm;
    const __bf16* ur = UIT + ((size_t)b * kD + n) * kC;
    acc[j] = wmma_bf16(aA0, frag_row(ur, khalf), acc[j]);
    acc[j] = wmma_bf16(aA1, frag_row(ur + 32, khalf), acc[j]);
  }
  // part 2: Q_t @ S (K = 1024, B rows from transposed S shadow)
  for (int kk = 0; kk < kD; kk += 32) {
    bf16x16 a = frag_row(Q + (rowBase + tmw * 16 + lm) * kD + kk, khalf);
    for (int j = 0; j < 4; ++j) {
      int n = n0 + (tn0 + j) * 16 + lm;
      acc[j] = wmma_bf16(a, frag_row(SBb + (size_t)n * kD + kk, khalf), acc[j]);
    }
  }
  for (int j = 0; j < 4; ++j) {
    int n = n0 + (tn0 + j) * 16 + lm;
    for (int r = 0; r < 8; ++r) {
      int m = tmw * 16 + r + 8 * khalf;
      OB[(rowBase + m) * kD + n] = (__bf16)acc[j][r];
    }
  }
}

// ---------------- scan step C: S += K_t^T @ u_i (transposed state, no LDS) ----------------
__global__ __launch_bounds__(256)
void k_stepC(const __bf16* __restrict__ KT, const __bf16* __restrict__ UIT,
             float* __restrict__ SfT, __bf16* __restrict__ SbT, int t) {
  const int tid = threadIdx.x, lane = tid & 31, w = tid >> 5;
  const int khalf = lane >> 4, lm = lane & 15;
  const int mb = blockIdx.x, nbk = blockIdx.y, b = blockIdx.z;
  const int m0 = mb * 64, n0 = nbk * 128;
  const int tmw = w & 3, tn0 = (w >> 2) * 4;

  const __bf16* karow = KT + ((size_t)b * kD + m0 + tmw * 16 + lm) * kL + (size_t)t * kC;
  bf16x16 a0 = frag_row(karow, khalf);
  bf16x16 a1 = frag_row(karow + 32, khalf);

  for (int j = 0; j < 4; ++j) {
    int n = n0 + (tn0 + j) * 16 + lm;
    const __bf16* ur = UIT + ((size_t)b * kD + n) * kC;
    f32x8 acc = fzero();
    acc = wmma_bf16(a0, frag_row(ur, khalf), acc);
    acc = wmma_bf16(a1, frag_row(ur + 32, khalf), acc);
    for (int r = 0; r < 8; ++r) {
      int m = m0 + tmw * 16 + r + 8 * khalf;
      size_t idx = (size_t)b * kD * kD + (size_t)n * kD + m;  // transposed layout
      float v = SfT[idx] + acc[r];
      SfT[idx] = v;
      SbT[idx] = (__bf16)v;
    }
  }
}

// ---------------- final projection: out = O @ proj^T + proj_b ----------------
__global__ __launch_bounds__(256)
void k_proj(const __bf16* __restrict__ OB, const __bf16* __restrict__ WP,
            const float* __restrict__ bias, float* __restrict__ out) {
  const int tid = threadIdx.x, lane = tid & 31, w = tid >> 5;
  const int khalf = lane >> 4, lm = lane & 15;
  const int n0 = blockIdx.x * 128, m0 = blockIdx.y * 128;
  const int wm = (w & 3) * 32, wn = (w >> 2) * 64;

  f32x8 acc[2][4];
  for (int i = 0; i < 2; ++i) for (int j = 0; j < 4; ++j) acc[i][j] = fzero();

  for (int kk = 0; kk < kD; kk += 32) {
    bf16x16 a[2], b[4];
    for (int i = 0; i < 2; ++i) {
      const __bf16* ar = OB + (size_t)(m0 + wm + i * 16 + lm) * kD + kk;
      __builtin_prefetch(ar + 128, 0, 3);
      a[i] = frag_row(ar, khalf);
    }
    for (int j = 0; j < 4; ++j)
      b[j] = frag_row(WP + (size_t)(n0 + wn + j * 16 + lm) * kD + kk, khalf);
    for (int i = 0; i < 2; ++i)
      for (int j = 0; j < 4; ++j)
        acc[i][j] = wmma_bf16(a[i], b[j], acc[i][j]);
  }

  for (int i = 0; i < 2; ++i) for (int j = 0; j < 4; ++j)
    for (int r = 0; r < 8; ++r) {
      int m = m0 + wm + i * 16 + r + 8 * khalf;
      int n = n0 + wn + j * 16 + lm;
      out[(size_t)m * kD + n] = acc[i][j][r] + bias[n];
    }
}

// ---------------- host orchestration ----------------
extern "C" void kernel_launch(void* const* d_in, const int* in_sizes, int n_in,
                              void* d_out, int out_size, void* d_ws, size_t ws_size,
                              hipStream_t stream) {
  (void)in_sizes; (void)n_in; (void)out_size; (void)ws_size;
  const float* X      = (const float*)d_in[0];
  const float* Wq_w   = (const float*)d_in[1];
  const float* Wq_b   = (const float*)d_in[2];
  const float* Wk_w   = (const float*)d_in[3];
  const float* Wk_b   = (const float*)d_in[4];
  const float* Wv_w   = (const float*)d_in[5];
  const float* Wv_b   = (const float*)d_in[6];
  const float* beta_w = (const float*)d_in[7];
  const float* beta_b = (const float*)d_in[8];
  const float* proj_w = (const float*)d_in[9];
  const float* proj_b = (const float*)d_in[10];
  float* out = (float*)d_out;

  char* p = (char*)d_ws;
  auto alloc = [&](size_t bytes) -> void* {
    void* r = (void*)p;
    p += (bytes + 255) & ~(size_t)255;
    return r;
  };

  const size_t MLD = (size_t)kML * kD;
  __bf16* XB   = (__bf16*)alloc(MLD * 2);
  __bf16* WQB  = (__bf16*)alloc((size_t)kD * kD * 2);
  __bf16* WKB  = (__bf16*)alloc((size_t)kD * kD * 2);
  __bf16* WVB  = (__bf16*)alloc((size_t)kD * kD * 2);
  __bf16* WPB  = (__bf16*)alloc((size_t)kD * kD * 2);
  float*  BETA = (float*)alloc((size_t)kML * 4);
  __bf16* QB   = (__bf16*)alloc(MLD * 2);
  __bf16* Krow = (__bf16*)alloc(MLD * 2);
  __bf16* Kbrw = (__bf16*)alloc(MLD * 2);
  __bf16* KT   = (__bf16*)alloc(MLD * 2);   // [b][d][l]
  __bf16* KbT  = (__bf16*)alloc(MLD * 2);   // [b][d][l]
  __bf16* VbT  = (__bf16*)alloc(MLD * 2);   // [b][d][l]
  __bf16* Wm   = (__bf16*)alloc(MLD * 2);
  __bf16* Um   = (__bf16*)alloc(MLD * 2);
  float*  SfT  = (float*)alloc((size_t)kB * kD * kD * 4);   // S^T, f32 master
  __bf16* SbT  = (__bf16*)alloc((size_t)kB * kD * kD * 2);  // S^T, bf16 shadow
  __bf16* UIT  = (__bf16*)alloc((size_t)kB * kD * kC * 2);  // u_i^T
  __bf16* ABuf = (__bf16*)alloc((size_t)kB * kC * kC * 2);
  __bf16* OB   = (__bf16*)alloc(MLD * 2);

  k_cvt_bf16<<<2048, 256, 0, stream>>>(X, XB, (int)MLD);
  k_cvt_bf16<<<1024, 256, 0, stream>>>(Wq_w, WQB, kD * kD);
  k_cvt_bf16<<<1024, 256, 0, stream>>>(Wk_w, WKB, kD * kD);
  k_cvt_bf16<<<1024, 256, 0, stream>>>(Wv_w, WVB, kD * kD);
  k_cvt_bf16<<<1024, 256, 0, stream>>>(proj_w, WPB, kD * kD);
  k_beta<<<kML / 8, 256, 0, stream>>>(X, beta_w, beta_b, BETA);
  k_zeroS<<<2048, 256, 0, stream>>>(SfT, SbT, kB * kD * kD);

  k_gemm_qkv<<<dim3(8, 128, 3), 256, 0, stream>>>(
      XB, WQB, WKB, WVB, Wq_b, Wk_b, Wv_b, BETA,
      QB, Krow, Kbrw, KT, KbT, VbT);

  k_chunkprep<<<kB * kN, 256, 0, stream>>>(Krow, Kbrw, KbT, VbT, Wm, Um);

  for (int t = 0; t < kN; ++t) {
    k_stepA<<<dim3(9, kB), 256, 0, stream>>>(Wm, Um, QB, Krow, SbT, UIT, ABuf, t);
    k_stepB<<<dim3(8, kB), 256, 0, stream>>>(ABuf, UIT, QB, SbT, OB, t);
    k_stepC<<<dim3(16, 8, kB), 256, 0, stream>>>(KT, UIT, SfT, SbT, t);
  }

  k_proj<<<dim3(8, 128), 256, 0, stream>>>(OB, WPB, proj_b, out);
}
